// CenterHeadInf_32538672235142
// MI455X (gfx1250) — compile-verified, matrix operand order checked by reading, not run on hardware
//
#include <hip/hip_runtime.h>
#include <hip/hip_bf16.h>
#include <math.h>

typedef __attribute__((ext_vector_type(16))) _Float16 v16h;
typedef __attribute__((ext_vector_type(8)))  _Float16 v8h;
typedef __attribute__((ext_vector_type(8)))  float    v8f;

#define HH 360
#define WW 360
#define HWSZ (HH * WW)
#define KTOP 500

// ---------------------------------------------------------------------------
// Weight repack: fold BN scale into weights and emit them in the exact
// per-lane V_WMMA_F32_16X16X32_F16 B-fragment order, so the conv hot loop
// reads one contiguous v16h per lane per fragment.
// Fragment order: frag = (tap*KB + kb)*NB + nb ; within fragment: lane*16 + e.
// Element mapping (wave32, 16-bit B 32x16):
//   n  = lane % 16
//   k  = kb*32 + (lane>=16 ? 8 : 0) + (e<8 ? e : e+8)
// ---------------------------------------------------------------------------
__global__ void wprep_kernel(const float* __restrict__ w,
                             const float* __restrict__ scale,
                             _Float16* __restrict__ dst,
                             int CI, int NB, int coutValid) {
  const int KB  = CI / 32;
  const int frag = blockIdx.x;                 // frag < 9*KB*NB
  const int nb  = frag % NB;
  const int kb  = (frag / NB) % KB;
  const int tap = frag / (NB * KB);
  const int lane = threadIdx.x;                // 32 threads
  const int co = nb * 16 + (lane & 15);
  const int kbase = kb * 32 + ((lane >> 4) ? 8 : 0);
  #pragma unroll
  for (int e = 0; e < 16; ++e) {
    const int off = (e < 8) ? e : (e + 8);
    const int ci = kbase + off;
    float v = 0.0f;
    if (co < coutValid) {
      v = w[((size_t)co * CI + ci) * 9 + tap];
      if (scale) v *= scale[co];
    }
    dst[(size_t)frag * 512 + lane * 16 + e] = (_Float16)v;
  }
}

// ---------------------------------------------------------------------------
// Implicit-GEMM 3x3 SAME conv via WMMA.
//   grid  = (3, 360): blockIdx.x -> 128-pixel x-strip, blockIdx.y -> row y
//   block = 256 (8 waves); wave w computes pixels [x0+16w, x0+16w+15] x 16*NB cout
// Input staged to LDS in 64-channel chunks as f16, channel-contiguous per
// pixel (stride CIP=72 halves -> 16B-aligned v8h A-fragment reads).
// ---------------------------------------------------------------------------
template<int CI, int NB, bool SRC_HALF, bool HALF_OUT>
__global__ __launch_bounds__(256)
void conv3x3_wmma(const void* __restrict__ srcv,
                  const _Float16* __restrict__ wfrag,
                  const float* __restrict__ bias,
                  void* __restrict__ dstv,
                  int coutStore) {
  constexpr int XT  = 130;   // x0-1 .. x0+128
  constexpr int CIP = 72;    // 64-channel chunk + 8-half pad (bank spread)
  alignas(16) __shared__ _Float16 tile[3 * XT * CIP];

  const int x0   = blockIdx.x * 128;
  const int y    = blockIdx.y;
  const int tid  = threadIdx.x;
  const int wave = tid >> 5;
  const int lane = tid & 31;
  const int m    = lane & 15;
  const int hi   = lane >> 4;
  const int KB   = CI / 32;

  const v8f vzero = {0.f, 0.f, 0.f, 0.f, 0.f, 0.f, 0.f, 0.f};
  v8f acc[NB];
  #pragma unroll
  for (int nb = 0; nb < NB; ++nb) acc[nb] = vzero;

  for (int ch = 0; ch < CI; ch += 64) {
    __syncthreads();
    // stage 64 channels x 3 rows x 130 px (zero-padded borders), f32/f16 -> f16
    for (int e = tid; e < 64 * 3 * XT; e += 256) {
      const int xi = e % XT;
      const int r  = (e / XT) % 3;
      const int ci = e / (3 * XT);
      const int gy = y + r - 1;
      const int gx = x0 + xi - 1;
      float v = 0.0f;
      if (gy >= 0 && gy < HH && gx >= 0 && gx < WW) {
        const size_t gidx = (size_t)(ch + ci) * HWSZ + (size_t)gy * WW + gx;
        v = SRC_HALF ? (float)((const _Float16*)srcv)[gidx]
                     : ((const float*)srcv)[gidx];
      }
      tile[(r * XT + xi) * CIP + ci] = (_Float16)v;
    }
    __syncthreads();

    #pragma unroll
    for (int tap = 0; tap < 9; ++tap) {
      const int ky = tap / 3;
      const int kx = tap % 3;
      #pragma unroll
      for (int kb = 0; kb < 2; ++kb) {          // two 32-ch K-blocks per chunk
        const int kbg = (ch >> 5) + kb;
        const int xi  = wave * 16 + m + kx;
        const _Float16* p = &tile[(ky * XT + xi) * CIP + kb * 32 + hi * 8];
        union { v16h v; v8h h[2]; } au;
        au.h[0] = *(const v8h*)p;               // K = base..base+7
        au.h[1] = *(const v8h*)(p + 16);        // K = base+16..base+23
        #pragma unroll
        for (int nb = 0; nb < NB; ++nb) {
          const _Float16* bp =
              &wfrag[((size_t)(tap * KB + kbg) * NB + nb) * 512 + lane * 16];
          __builtin_prefetch(bp + 512, 0, 1);
          const v16h b = *(const v16h*)bp;
          acc[nb] = __builtin_amdgcn_wmma_f32_16x16x32_f16(
              false, au.v, false, b, (short)0, acc[nb], false, false);
        }
      }
    }
  }

  // Epilogue: D layout -> VGPR r, lane l: pixel = 16*wave + r + 8*hi, co = nb*16 + (l&15)
  const int pxb = x0 + wave * 16 + hi * 8;
  union { v8f v; float f[8]; } cu;
  #pragma unroll
  for (int nb = 0; nb < NB; ++nb) {
    cu.v = acc[nb];
    const int co = nb * 16 + (lane & 15);
    #pragma unroll
    for (int r = 0; r < 8; ++r) {
      const int px = pxb + r;
      if (px < WW) {
        if (HALF_OUT) {
          float v = cu.f[r] + bias[co];
          v = v > 0.f ? v : 0.f;
          ((_Float16*)dstv)[(size_t)co * HWSZ + (size_t)y * WW + px] = (_Float16)v;
        } else if (co < coutStore) {
          const float v = cu.f[r] + bias[co];
          ((float*)dstv)[(size_t)co * HWSZ + (size_t)y * WW + px] = v;
        }
      }
    }
  }
}

// ---------------------------------------------------------------------------
// Post-processing
// ---------------------------------------------------------------------------
__global__ void sigmoid_kernel(const float* __restrict__ hm,
                               float* __restrict__ sc, int n) {
  const int i = blockIdx.x * 256 + threadIdx.x;
  if (i < n) sc[i] = 1.0f / (1.0f + expf(-hm[i]));
}

__global__ __launch_bounds__(1024)
void topk_kernel(float* __restrict__ sc, int* __restrict__ outIdx,
                 float* __restrict__ outSc) {
  __shared__ float sk[1024];
  __shared__ int   si[1024];
  const int N = 3 * HWSZ;
  for (int it = 0; it < KTOP; ++it) {
    float best = -1e30f;
    int   bi   = 0x7fffffff;
    for (int i = threadIdx.x; i < N; i += 1024) {
      const float v = sc[i];
      if (v > best) { best = v; bi = i; }
    }
    sk[threadIdx.x] = best;
    si[threadIdx.x] = bi;
    __syncthreads();
    for (int s = 512; s > 0; s >>= 1) {
      if (threadIdx.x < s) {
        const float ov = sk[threadIdx.x + s];
        const int   oi = si[threadIdx.x + s];
        if (ov > sk[threadIdx.x] ||
            (ov == sk[threadIdx.x] && oi < si[threadIdx.x])) {
          sk[threadIdx.x] = ov;
          si[threadIdx.x] = oi;
        }
      }
      __syncthreads();
    }
    if (threadIdx.x == 0) {
      outIdx[it] = si[0];
      outSc[it]  = sk[0];
      sc[si[0]]  = -1e30f;
    }
    __syncthreads();
  }
}

__global__ void decode_kernel(const int* __restrict__ topIdx,
                              const float* __restrict__ topSc,
                              const float* __restrict__ ct,
                              const float* __restrict__ cz,
                              const float* __restrict__ dm,
                              const float* __restrict__ rt,
                              const float* __restrict__ io,
                              float* __restrict__ boxes,
                              float* __restrict__ scoreK,
                              int* __restrict__ labelK,
                              int* __restrict__ validK) {
  const int k = blockIdx.x * 256 + threadIdx.x;
  if (k >= KTOP) return;
  const int idx   = topIdx[k];
  const int label = idx / HWSZ;
  const int sp    = idx % HWSZ;
  float xs = (float)(sp % WW);
  float ys = (float)(sp / WW);
  xs = (xs + ct[sp])        * 0.32f - 57.6f;
  ys = (ys + ct[HWSZ + sp]) * 0.32f - 57.6f;
  const float z   = cz[sp];
  const float d0  = expf(dm[sp]);
  const float d1  = expf(dm[HWSZ + sp]);
  const float d2  = expf(dm[2 * HWSZ + sp]);
  const float ang = atan2f(rt[HWSZ + sp], rt[sp]);
  float iov = (io[sp] + 1.0f) * 0.5f;
  iov = fminf(fmaxf(iov, 0.0f), 1.0f);
  const float rect[3] = {0.68f, 0.71f, 0.65f};
  const float r = rect[label];
  float s = powf(topSc[k], 1.0f - r) * powf(iov, r);
  const bool in_range = xs >= -61.2f && xs <= 61.2f &&
                        ys >= -61.2f && ys <= 61.2f &&
                        z  >= -10.0f && z  <= 10.0f;
  const bool valid = (s > 0.1f) && in_range;
  s = valid ? s : 0.0f;
  float* b = boxes + (size_t)k * 7;
  b[0] = xs; b[1] = ys; b[2] = z; b[3] = d0; b[4] = d1; b[5] = d2; b[6] = ang;
  scoreK[k] = s;
  labelK[k] = label;
  validK[k] = valid ? 1 : 0;
}

__global__ __launch_bounds__(512)
void sort_kernel(const float* __restrict__ boxes, const float* __restrict__ scoreK,
                 const int* __restrict__ labelK, const int* __restrict__ validK,
                 float* __restrict__ boxesS, float* __restrict__ scoreS,
                 int* __restrict__ labelS, int* __restrict__ validS) {
  __shared__ float key[512];
  __shared__ int   val[512];
  const int t = threadIdx.x;
  key[t] = (t < KTOP) ? scoreK[t] : -1e30f;
  val[t] = t;
  __syncthreads();
  for (int size = 2; size <= 512; size <<= 1) {
    for (int stride = size >> 1; stride > 0; stride >>= 1) {
      const int partner = t ^ stride;
      if (partner > t) {
        const bool desc = ((t & size) == 0);
        const float k0 = key[t], k1 = key[partner];
        const int   v0 = val[t], v1 = val[partner];
        const bool sw = desc ? (k0 < k1 || (k0 == k1 && v0 > v1))
                             : (k0 > k1 || (k0 == k1 && v0 < v1));
        if (sw) { key[t] = k1; key[partner] = k0; val[t] = v1; val[partner] = v0; }
      }
      __syncthreads();
    }
  }
  if (t < KTOP) {
    const int src = val[t];
    if (src < KTOP) {
      for (int c = 0; c < 7; ++c) boxesS[t * 7 + c] = boxes[src * 7 + c];
      scoreS[t] = scoreK[src];
      labelS[t] = labelK[src];
      validS[t] = validK[src];
    } else {
      for (int c = 0; c < 7; ++c) boxesS[t * 7 + c] = 0.f;
      scoreS[t] = 0.f; labelS[t] = 0; validS[t] = 0;
    }
  }
}

__global__ __launch_bounds__(512)
void nms_kernel(const float* __restrict__ boxesS, const int* __restrict__ validS,
                int* __restrict__ keep) {
  __shared__ float x1[512], x2[512], y1[512], y2[512], ar[512];
  __shared__ int kp[512];
  const int t = threadIdx.x;
  float x = 0.f, yy = 0.f, dx = 0.f, dy = 0.f;
  if (t < KTOP) {
    x  = boxesS[t * 7 + 0];
    yy = boxesS[t * 7 + 1];
    dx = boxesS[t * 7 + 3];
    dy = boxesS[t * 7 + 4];
  }
  x1[t] = x - dx * 0.5f;  x2[t] = x + dx * 0.5f;
  y1[t] = yy - dy * 0.5f; y2[t] = yy + dy * 0.5f;
  ar[t] = dx * dy;
  kp[t] = (t < KTOP) ? validS[t] : 0;
  __syncthreads();
  for (int i = 0; i < KTOP; ++i) {
    const bool ki = kp[i] != 0;
    __syncthreads();
    if (ki && t > i && t < KTOP) {
      float ix = fminf(x2[t], x2[i]) - fmaxf(x1[t], x1[i]);
      float iy = fminf(y2[t], y2[i]) - fmaxf(y1[t], y1[i]);
      ix = fmaxf(ix, 0.f); iy = fmaxf(iy, 0.f);
      const float inter = ix * iy;
      const float iou = inter / fmaxf(ar[t] + ar[i] - inter, 1e-6f);
      if (iou > 0.7f) kp[t] = 0;
    }
    __syncthreads();
  }
  if (t < KTOP) keep[t] = kp[t];
}

__global__ void finalize_kernel(const float* __restrict__ boxesS,
                                const float* __restrict__ scoreS,
                                const int* __restrict__ labelS,
                                const int* __restrict__ keep,
                                float* __restrict__ out) {
  const int t = blockIdx.x * 256 + threadIdx.x;
  if (t < KTOP * 7) out[t] = boxesS[t];
  if (t < KTOP) {
    out[3500 + t] = scoreS[t] * (float)keep[t];
    out[4000 + t] = (float)labelS[t];
    out[4500 + t] = (float)keep[t];
  }
}

// ---------------------------------------------------------------------------
// Host-side orchestration
// ---------------------------------------------------------------------------
extern "C" void kernel_launch(void* const* d_in, const int* in_sizes, int n_in,
                              void* d_out, int out_size, void* d_ws, size_t ws_size,
                              hipStream_t stream) {
  (void)in_sizes; (void)n_in; (void)out_size; (void)ws_size;

  const float* x        = (const float*)d_in[0];
  const float* w_shared = (const float*)d_in[1];
  const float* s_shared = (const float*)d_in[2];
  const float* b_shared = (const float*)d_in[3];
  const float *w0[6], *s0[6], *b0[6], *w1[6], *b1[6];
  const int couts[6] = {3, 2, 1, 3, 2, 1};    // hm, ct, cz, dim, rot, iou
  for (int i = 0; i < 6; ++i) {
    const int base = 4 + 5 * i;
    w0[i] = (const float*)d_in[base + 0];
    s0[i] = (const float*)d_in[base + 1];
    b0[i] = (const float*)d_in[base + 2];
    w1[i] = (const float*)d_in[base + 3];
    b1[i] = (const float*)d_in[base + 4];
  }

  char* ws = (char*)d_ws;
  size_t off = 0;
  auto alloc = [&](size_t bytes) -> void* {
    off = (off + 255) & ~(size_t)255;
    void* p = ws + off;
    off += bytes;
    return p;
  };

  _Float16* sharedF16 = (_Float16*)alloc((size_t)64 * HWSZ * 2);
  _Float16* hF16      = (_Float16*)alloc((size_t)64 * HWSZ * 2);
  float* bout[6];
  for (int i = 0; i < 6; ++i) bout[i] = (float*)alloc((size_t)couts[i] * HWSZ * 4);
  _Float16* wfS = (_Float16*)alloc((size_t)144 * 512 * 2);
  _Float16* wf0[6];
  _Float16* wf1[6];
  for (int i = 0; i < 6; ++i) wf0[i] = (_Float16*)alloc((size_t)72 * 512 * 2);
  for (int i = 0; i < 6; ++i) wf1[i] = (_Float16*)alloc((size_t)18 * 512 * 2);
  float* sc     = (float*)alloc((size_t)3 * HWSZ * 4);
  int*   topIdx = (int*)alloc(KTOP * 4);
  float* topSc  = (float*)alloc(KTOP * 4);
  float* boxes  = (float*)alloc(KTOP * 7 * 4);
  float* scoreK = (float*)alloc(KTOP * 4);
  int*   labelK = (int*)alloc(KTOP * 4);
  int*   validK = (int*)alloc(KTOP * 4);
  float* boxesS = (float*)alloc(KTOP * 7 * 4);
  float* scoreS = (float*)alloc(KTOP * 4);
  int*   labelS = (int*)alloc(KTOP * 4);
  int*   validS = (int*)alloc(KTOP * 4);
  int*   keep   = (int*)alloc(KTOP * 4);

  // Weight repack (fold BN scale, B-fragment order)
  wprep_kernel<<<9 * 4 * 4, 32, 0, stream>>>(w_shared, s_shared, wfS, 128, 4, 64);
  for (int i = 0; i < 6; ++i) {
    wprep_kernel<<<9 * 2 * 4, 32, 0, stream>>>(w0[i], s0[i], wf0[i], 64, 4, 64);
    wprep_kernel<<<9 * 2 * 1, 32, 0, stream>>>(w1[i], nullptr, wf1[i], 64, 1, couts[i]);
  }

  const dim3 cgrid(3, HH);
  // shared = relu(conv(x) * s + b), f16
  conv3x3_wmma<128, 4, false, true><<<cgrid, 256, 0, stream>>>(
      x, wfS, b_shared, sharedF16, 64);
  // per-branch: h = relu(conv(shared)*s+b) (f16), head = conv(h)+b (f32)
  for (int i = 0; i < 6; ++i) {
    conv3x3_wmma<64, 4, true, true><<<cgrid, 256, 0, stream>>>(
        sharedF16, wf0[i], b0[i], hF16, 64);
    conv3x3_wmma<64, 1, true, false><<<cgrid, 256, 0, stream>>>(
        hF16, wf1[i], b1[i], bout[i], couts[i]);
  }

  sigmoid_kernel<<<(3 * HWSZ + 255) / 256, 256, 0, stream>>>(bout[0], sc, 3 * HWSZ);
  topk_kernel<<<1, 1024, 0, stream>>>(sc, topIdx, topSc);
  decode_kernel<<<2, 256, 0, stream>>>(topIdx, topSc, bout[1], bout[2], bout[3],
                                       bout[4], bout[5], boxes, scoreK, labelK, validK);
  sort_kernel<<<1, 512, 0, stream>>>(boxes, scoreK, labelK, validK,
                                     boxesS, scoreS, labelS, validS);
  nms_kernel<<<1, 512, 0, stream>>>(boxesS, validS, keep);
  finalize_kernel<<<14, 256, 0, stream>>>(boxesS, scoreS, labelS, keep, (float*)d_out);
}
